// DecoderBlock_52493090291724
// MI455X (gfx1250) — matrix operand order repaired and checked by reading, not compiled, not run on hardware
//
#include <hip/hip_runtime.h>

// ---------------------------------------------------------------------------
// Transformer decoder block for MI455X (gfx1250, wave32, WMMA).
// B=4, T=TE=1024, C=1024, H=16, HD=64.  All matmuls in bf16 WMMA, f32 accum.
// ---------------------------------------------------------------------------

#define Bsz 4
#define Tsz 1024
#define Csz 1024
#define Hn  16
#define HDs 64

typedef unsigned short u16;
typedef __attribute__((ext_vector_type(16))) __bf16 v16bf;
typedef __attribute__((ext_vector_type(8)))  float  v8f;

union FragU { uint4 q[2]; v16bf f; };

__device__ __forceinline__ u16 f2bf(float f) {
  unsigned int u = __float_as_uint(f);
  unsigned int r = u + 0x7fffu + ((u >> 16) & 1u);   // round-to-nearest-even
  return (u16)(r >> 16);
}

__device__ __forceinline__ v8f v8f_zero() {
  v8f z = {0.f, 0.f, 0.f, 0.f, 0.f, 0.f, 0.f, 0.f};
  return z;
}

__device__ __forceinline__ v8f wmma_bf16(v16bf a, v16bf b, v8f c) {
  return __builtin_amdgcn_wmma_f32_16x16x32_bf16(
      /*neg_a=*/false, a, /*neg_b=*/false, b,
      /*c_mod=*/(short)0, c, /*reuse_a=*/false, /*reuse_b=*/false);
}

// A-operand fragment (16x32 bf16): lane<16 -> row=lane, K 0..7 & 16..23;
// lane>=16 -> row=lane-16, K 8..15 & 24..31. Two 16B LDS loads.
__device__ __forceinline__ v16bf load_a_frag(const u16* sm, int row_base, int k0,
                                             int lane, int stride) {
  int r    = row_base + (lane & 15);
  int koff = k0 + ((lane & 16) ? 8 : 0);
  const uint4* p = (const uint4*)(sm + r * stride + koff);
  FragU u;
  u.q[0] = p[0];
  u.q[1] = p[2];   // +16 elements = +32B
  return u.f;
}

// B-operand fragment (32x16 bf16, source stored as Bt[N][K]): lane<16 ->
// col=lane, K 0..7 & 16..23; lane>=16 -> col=lane-16, K 8..15 & 24..31.
__device__ __forceinline__ v16bf load_b_frag(const u16* sm, int n_base, int k0,
                                             int lane, int stride) {
  int n    = n_base + (lane & 15);
  int koff = k0 + ((lane & 16) ? 8 : 0);
  const uint4* p = (const uint4*)(sm + n * stride + koff);
  FragU u;
  u.q[0] = p[0];
  u.q[1] = p[2];
  return u.f;
}

// ---------------------------------------------------------------------------
// Weight packing: fp32 -> bf16, transposed to [N][K] (K = input dim contiguous)
// ---------------------------------------------------------------------------

// W[H, C, HD] -> Wt[(h*HD+d)*C + c] * scale   (scale folds 1/sqrt(HD) into Wq)
__global__ void pack_qkv_kernel(const float* __restrict__ w, u16* __restrict__ wt,
                                float scale) {
  int idx = blockIdx.x * 256 + threadIdx.x;      // over C*C = 1M
  int n = idx >> 10, c = idx & 1023;
  int h = n >> 6, d = n & 63;
  wt[idx] = f2bf(w[(h * Csz + c) * HDs + d] * scale);
}

// W[C_in, C_out] -> Wt[n*C + c]  (transpose)
__global__ void pack_mat_kernel(const float* __restrict__ w, u16* __restrict__ wt) {
  int idx = blockIdx.x * 256 + threadIdx.x;
  int n = idx >> 10, c = idx & 1023;
  wt[idx] = f2bf(w[c * Csz + n]);
}

// ---------------------------------------------------------------------------
// LayerNorm: fp32 in -> bf16 out.  One block (256 thr) per row of 1024.
// ---------------------------------------------------------------------------
__global__ __launch_bounds__(256)
void ln_bf16_kernel(const float* __restrict__ x, const float* __restrict__ g,
                    const float* __restrict__ be, u16* __restrict__ out) {
  __shared__ float red[8];
  __shared__ float bcast;
  int row = blockIdx.x, tid = threadIdx.x;
  const float* xr = x + (size_t)row * Csz;
  float v[4];
  float s = 0.f;
#pragma unroll
  for (int i = 0; i < 4; i++) { v[i] = xr[tid + i * 256]; s += v[i]; }
  int lane = tid & 31, wid = tid >> 5;
#pragma unroll
  for (int off = 16; off > 0; off >>= 1) s += __shfl_xor(s, off, 32);
  if (lane == 0) red[wid] = s;
  __syncthreads();
  if (tid == 0) { float t = 0; for (int i = 0; i < 8; i++) t += red[i]; bcast = t * (1.f / 1024.f); }
  __syncthreads();
  float mean = bcast;
  __syncthreads();
  s = 0.f;
#pragma unroll
  for (int i = 0; i < 4; i++) { float d = v[i] - mean; s += d * d; }
#pragma unroll
  for (int off = 16; off > 0; off >>= 1) s += __shfl_xor(s, off, 32);
  if (lane == 0) red[wid] = s;
  __syncthreads();
  if (tid == 0) { float t = 0; for (int i = 0; i < 8; i++) t += red[i]; bcast = t * (1.f / 1024.f); }
  __syncthreads();
  float rstd = rsqrtf(bcast + 1e-5f);
#pragma unroll
  for (int i = 0; i < 4; i++) {
    int c = tid + i * 256;
    out[(size_t)row * Csz + c] = f2bf((v[i] - mean) * rstd * g[c] + be[c]);
  }
}

// ---------------------------------------------------------------------------
// bf16 GEMM: out[M,N] = A[M,K] * Bt[N,K]^T with fused, compile-time epilogue.
// Block tile 128x128, 8 waves each 64x32, K-step 32, LDS double-buffered.
//   MODE 0: outbf = acc                       (QKV projections)
//   MODE 1: out32 = acc + bias + resid        (attn proj / final FFN)
//   MODE 2: outbf = gelu(acc + bias)          (FFN hidden)
// ---------------------------------------------------------------------------
template <int MODE>
__global__ __launch_bounds__(256)
__attribute__((amdgpu_waves_per_eu(1)))
void gemm_bf16_kernel(const u16* __restrict__ A, const u16* __restrict__ Bt,
                      const float* __restrict__ bias, const float* __restrict__ resid,
                      float* __restrict__ out32, u16* __restrict__ outbf,
                      int M, int N, int K) {
  __shared__ u16 As[2][128 * 40];
  __shared__ u16 Bs[2][128 * 40];
  int tid  = threadIdx.x;
  int lane = tid & 31, w = tid >> 5;
  int wr = w >> 2, wc = w & 3;            // 2x4 wave grid
  int m0 = blockIdx.y * 128, n0 = blockIdx.x * 128;

  // staging registers for global->LDS double buffering (2 chunks each)
  int r0 = tid >> 2, kc0 = (tid & 3) * 8;           // chunk 0: rows 0..63
  int r1 = r0 + 64, kc1 = kc0;                      // chunk 1: rows 64..127
  uint4 ra0, ra1, rb0, rb1;

  v8f acc[4][2];
#pragma unroll
  for (int i = 0; i < 4; i++)
#pragma unroll
    for (int j = 0; j < 2; j++) acc[i][j] = v8f_zero();

  // prologue: fetch tile 0
  ra0 = *(const uint4*)&A [(size_t)(m0 + r0) * K + kc0];
  ra1 = *(const uint4*)&A [(size_t)(m0 + r1) * K + kc1];
  rb0 = *(const uint4*)&Bt[(size_t)(n0 + r0) * K + kc0];
  rb1 = *(const uint4*)&Bt[(size_t)(n0 + r1) * K + kc1];
  *(uint4*)&As[0][r0 * 40 + kc0] = ra0;
  *(uint4*)&As[0][r1 * 40 + kc1] = ra1;
  *(uint4*)&Bs[0][r0 * 40 + kc0] = rb0;
  *(uint4*)&Bs[0][r1 * 40 + kc1] = rb1;

  const int NT = K / 32;
  for (int kt = 0; kt < NT; kt++) {
    __syncthreads();                       // buffer kt&1 is ready
    if (kt + 1 < NT) {                     // issue next-tile global loads early
      int k0 = (kt + 1) * 32;
      ra0 = *(const uint4*)&A [(size_t)(m0 + r0) * K + k0 + kc0];
      ra1 = *(const uint4*)&A [(size_t)(m0 + r1) * K + k0 + kc1];
      rb0 = *(const uint4*)&Bt[(size_t)(n0 + r0) * K + k0 + kc0];
      rb1 = *(const uint4*)&Bt[(size_t)(n0 + r1) * K + k0 + kc1];
    }
    int cur = kt & 1;
    v16bf af[4], bf[2];
#pragma unroll
    for (int i = 0; i < 4; i++) af[i] = load_a_frag(As[cur], wr * 64 + i * 16, 0, lane, 40);
#pragma unroll
    for (int j = 0; j < 2; j++) bf[j] = load_b_frag(Bs[cur], wc * 32 + j * 16, 0, lane, 40);
#pragma unroll
    for (int i = 0; i < 4; i++)
#pragma unroll
      for (int j = 0; j < 2; j++)
        acc[i][j] = wmma_bf16(af[i], bf[j], acc[i][j]);
    if (kt + 1 < NT) {                     // park next tile in the other buffer
      int nxt = cur ^ 1;
      *(uint4*)&As[nxt][r0 * 40 + kc0] = ra0;
      *(uint4*)&As[nxt][r1 * 40 + kc1] = ra1;
      *(uint4*)&Bs[nxt][r0 * 40 + kc0] = rb0;
      *(uint4*)&Bs[nxt][r1 * 40 + kc1] = rb1;
    }
  }

  int rsel = (lane & 16) ? 8 : 0;
#pragma unroll
  for (int i = 0; i < 4; i++)
#pragma unroll
    for (int v = 0; v < 8; v++) {
      int row = m0 + wr * 64 + i * 16 + v + rsel;
      size_t rb = (size_t)row * N;
#pragma unroll
      for (int j = 0; j < 2; j++) {
        int col = n0 + wc * 32 + j * 16 + (lane & 15);
        float val = acc[i][j][v];
        if (MODE == 0) {
          outbf[rb + col] = f2bf(val);
        } else if (MODE == 1) {
          val += bias[col] + resid[rb + col];
          out32[rb + col] = val;
        } else {                            // MODE == 2
          val += bias[col];
          val = 0.5f * val * (1.f + erff(val * 0.70710678f));
          outbf[rb + col] = f2bf(val);
        }
      }
    }
}

// ---------------------------------------------------------------------------
// Flash attention (bf16 in/out).  One block per (b, h, 128-row q tile).
// Key blocks of 64; online softmax; all matmuls via WMMA.
// The 1/sqrt(HD) scale is pre-folded into the Q projection weights.
// ---------------------------------------------------------------------------
__global__ __launch_bounds__(256)
__attribute__((amdgpu_waves_per_eu(1)))
void attn_kernel(const u16* __restrict__ Qp, const u16* __restrict__ Kp,
                 const u16* __restrict__ Vp, u16* __restrict__ Out,
                 int S, int causal) {
  __shared__ u16 Qs[128 * 72];
  __shared__ u16 Ks[64 * 72];
  __shared__ u16 Vt[64 * 72];   // transposed: Vt[d][s]
  __shared__ u16 Ps[128 * 72];  // per-wave-private rows

  int tid  = threadIdx.x;
  int lane = tid & 31, w = tid >> 5;
  int q0 = blockIdx.x * 128;
  int bh = blockIdx.y;
  int b = bh >> 4, h = bh & 15;
  const float NEG_INF = -__builtin_inff();

  size_t qbase = ((size_t)b * Tsz + q0) * Csz + h * HDs;
#pragma unroll
  for (int i = 0; i < 4; i++) {            // 128x64 Q tile, 1024 chunks
    int c = tid + i * 256;
    int r = c >> 3, kc = (c & 7) * 8;
    *(uint4*)&Qs[r * 72 + kc] = *(const uint4*)&Qp[qbase + (size_t)r * Csz + kc];
  }

  float mj[8], lj[8];
  v8f o[4];
#pragma unroll
  for (int j = 0; j < 8; j++) { mj[j] = NEG_INF; lj[j] = 0.f; }
#pragma unroll
  for (int t = 0; t < 4; t++) o[t] = v8f_zero();
  int rsel = (lane & 16) ? 8 : 0;

  int s_end = causal ? (q0 + 128) : S;
  if (s_end > S) s_end = S;

  for (int s0 = 0; s0 < s_end; s0 += 64) {
    __syncthreads();
    size_t kvbase = ((size_t)b * S + s0) * Csz + h * HDs;
#pragma unroll
    for (int i = 0; i < 2; i++) {          // 64x64 K and V tiles, 512 chunks
      int c = tid + i * 256;
      int r = c >> 3, kc = (c & 7) * 8;
      *(uint4*)&Ks[r * 72 + kc] = *(const uint4*)&Kp[kvbase + (size_t)r * Csz + kc];
      uint4 vv = *(const uint4*)&Vp[kvbase + (size_t)r * Csz + kc];
      const u16* pv = (const u16*)&vv;
#pragma unroll
      for (int e = 0; e < 8; e++) Vt[(kc + e) * 72 + r] = pv[e];
    }
    __syncthreads();

    // S = Q K^T  (contract over HD=64: two 32-wide WMMA steps)
    v16bf qa0 = load_a_frag(Qs, w * 16, 0,  lane, 72);
    v16bf qa1 = load_a_frag(Qs, w * 16, 32, lane, 72);
    v8f st[4];
#pragma unroll
    for (int t = 0; t < 4; t++) {
      st[t] = v8f_zero();
      st[t] = wmma_bf16(qa0, load_b_frag(Ks, t * 16, 0,  lane, 72), st[t]);
      st[t] = wmma_bf16(qa1, load_b_frag(Ks, t * 16, 32, lane, 72), st[t]);
    }

    // online softmax, row-wise (row lives across 16 lanes in C-layout)
#pragma unroll
    for (int j = 0; j < 8; j++) {
      int qg = q0 + w * 16 + j + rsel;
      float vt[4];
      float mx = NEG_INF;
#pragma unroll
      for (int t = 0; t < 4; t++) {
        float sv = st[t][j];
        if (causal && (s0 + t * 16 + (lane & 15)) > qg) sv = NEG_INF;
        vt[t] = sv;
        mx = fmaxf(mx, sv);
      }
#pragma unroll
      for (int off = 1; off < 16; off <<= 1) mx = fmaxf(mx, __shfl_xor(mx, off, 32));
      float mnew  = fmaxf(mj[j], mx);
      float alpha = __expf(mj[j] - mnew);
      float sum = 0.f;
      int prow = (w * 16 + j + rsel) * 72;
#pragma unroll
      for (int t = 0; t < 4; t++) {
        float p = __expf(vt[t] - mnew);
        sum += p;
        Ps[prow + t * 16 + (lane & 15)] = f2bf(p);
      }
#pragma unroll
      for (int off = 1; off < 16; off <<= 1) sum += __shfl_xor(sum, off, 32);
      lj[j] = lj[j] * alpha + sum;
      mj[j] = mnew;
#pragma unroll
      for (int dt = 0; dt < 4; dt++) o[dt][j] *= alpha;
    }

    // O += P V   (contract over 64 keys: two 32-wide WMMA steps)
    v16bf pa0 = load_a_frag(Ps, w * 16, 0,  lane, 72);
    v16bf pa1 = load_a_frag(Ps, w * 16, 32, lane, 72);
#pragma unroll
    for (int dt = 0; dt < 4; dt++) {
      o[dt] = wmma_bf16(pa0, load_b_frag(Vt, dt * 16, 0,  lane, 72), o[dt]);
      o[dt] = wmma_bf16(pa1, load_b_frag(Vt, dt * 16, 32, lane, 72), o[dt]);
    }
  }

#pragma unroll
  for (int dt = 0; dt < 4; dt++)
#pragma unroll
    for (int j = 0; j < 8; j++) {
      float val = o[dt][j] / lj[j];
      int row = q0 + w * 16 + j + rsel;
      int col = h * HDs + dt * 16 + (lane & 15);
      Out[((size_t)b * Tsz + row) * Csz + col] = f2bf(val);
    }
}

// ---------------------------------------------------------------------------
// Host orchestration
// ---------------------------------------------------------------------------
extern "C" void kernel_launch(void* const* d_in, const int* in_sizes, int n_in,
                              void* d_out, int out_size, void* d_ws, size_t ws_size,
                              hipStream_t stream) {
  const float* x_enc = (const float*)d_in[0];
  const float* x     = (const float*)d_in[1];
  const float* Wq_sa = (const float*)d_in[2];
  const float* Wk_sa = (const float*)d_in[3];
  const float* Wv_sa = (const float*)d_in[4];
  const float* Wp_sa = (const float*)d_in[5];
  const float* bp_sa = (const float*)d_in[6];
  const float* Wq_ca = (const float*)d_in[7];
  const float* Wk_ca = (const float*)d_in[8];
  const float* Wv_ca = (const float*)d_in[9];
  const float* Wp_ca = (const float*)d_in[10];
  const float* bp_ca = (const float*)d_in[11];
  const float* W1    = (const float*)d_in[12];
  const float* b1    = (const float*)d_in[13];
  const float* W2    = (const float*)d_in[14];
  const float* b2    = (const float*)d_in[15];
  const float* g1 = (const float*)d_in[16]; const float* be1 = (const float*)d_in[17];
  const float* g2 = (const float*)d_in[18]; const float* be2 = (const float*)d_in[19];
  const float* g3 = (const float*)d_in[20]; const float* be3 = (const float*)d_in[21];
  const float* g4 = (const float*)d_in[22]; const float* be4 = (const float*)d_in[23];

  char* ws = (char*)d_ws;
  size_t off = 0;
  auto take = [&](size_t bytes) -> void* {
    void* p = ws + off;
    off += (bytes + 255) & ~(size_t)255;
    return p;
  };

  const size_t Wbytes = (size_t)Csz * Csz * sizeof(u16);            // 2 MB
  const size_t Abytes = (size_t)Bsz * Tsz * Csz * sizeof(u16);      // 8 MB
  const size_t Fbytes = (size_t)Bsz * Tsz * Csz * sizeof(float);    // 16 MB

  u16* wtq_sa = (u16*)take(Wbytes);
  u16* wtk_sa = (u16*)take(Wbytes);
  u16* wtv_sa = (u16*)take(Wbytes);
  u16* wtp_sa = (u16*)take(Wbytes);
  u16* wtq_ca = (u16*)take(Wbytes);
  u16* wtk_ca = (u16*)take(Wbytes);
  u16* wtv_ca = (u16*)take(Wbytes);
  u16* wtp_ca = (u16*)take(Wbytes);
  u16* wt1    = (u16*)take(Wbytes);
  u16* wt2    = (u16*)take(Wbytes);
  u16* XN  = (u16*)take(Abytes);   // LN output (reused for ln1/ln3/ln4)
  u16* XE  = (u16*)take(Abytes);   // LN of encoder
  u16* Qb  = (u16*)take(Abytes);
  u16* Kb  = (u16*)take(Abytes);
  u16* Vb  = (u16*)take(Abytes);
  u16* ATT = (u16*)take(Abytes);
  u16* Hb  = (u16*)take(Abytes);   // GELU(ffn hidden)
  float* X1 = (float*)take(Fbytes);
  float* X2 = (float*)take(Fbytes);
  (void)ws_size; (void)in_sizes; (void)n_in; (void)out_size;

  const int M = Bsz * Tsz;             // 4096
  dim3 gGrid(Csz / 128, M / 128);      // (8, 32)
  dim3 aGrid(Tsz / 128, Bsz * Hn);     // (8, 64)
  const int packGrid = (Csz * Csz) / 256;
  const float qscale = 0.125f;         // 1/sqrt(HD)

  auto gemm_plain = [&](const u16* A, const u16* Bt, u16* obf) {
    gemm_bf16_kernel<0><<<gGrid, 256, 0, stream>>>(A, Bt, nullptr, nullptr,
                                                   nullptr, obf, M, Csz, Csz);
  };
  auto gemm_proj = [&](const u16* A, const u16* Bt, const float* bias,
                       const float* resid, float* o32) {
    gemm_bf16_kernel<1><<<gGrid, 256, 0, stream>>>(A, Bt, bias, resid,
                                                   o32, nullptr, M, Csz, Csz);
  };
  auto gemm_gelu = [&](const u16* A, const u16* Bt, const float* bias, u16* obf) {
    gemm_bf16_kernel<2><<<gGrid, 256, 0, stream>>>(A, Bt, bias, nullptr,
                                                   nullptr, obf, M, Csz, Csz);
  };

  // --- pack weights (bf16, [N][K]); 1/sqrt(HD) folded into Wq ---
  pack_qkv_kernel<<<packGrid, 256, 0, stream>>>(Wq_sa, wtq_sa, qscale);
  pack_qkv_kernel<<<packGrid, 256, 0, stream>>>(Wk_sa, wtk_sa, 1.f);
  pack_qkv_kernel<<<packGrid, 256, 0, stream>>>(Wv_sa, wtv_sa, 1.f);
  pack_mat_kernel<<<packGrid, 256, 0, stream>>>(Wp_sa, wtp_sa);
  pack_qkv_kernel<<<packGrid, 256, 0, stream>>>(Wq_ca, wtq_ca, qscale);
  pack_qkv_kernel<<<packGrid, 256, 0, stream>>>(Wk_ca, wtk_ca, 1.f);
  pack_qkv_kernel<<<packGrid, 256, 0, stream>>>(Wv_ca, wtv_ca, 1.f);
  pack_mat_kernel<<<packGrid, 256, 0, stream>>>(Wp_ca, wtp_ca);
  pack_mat_kernel<<<packGrid, 256, 0, stream>>>(W1, wt1);
  pack_mat_kernel<<<packGrid, 256, 0, stream>>>(W2, wt2);

  // --- self-attention: x1 = x + proj(attn(ln1(x))) ---
  ln_bf16_kernel<<<M, 256, 0, stream>>>(x, g1, be1, XN);
  gemm_plain(XN, wtq_sa, Qb);
  gemm_plain(XN, wtk_sa, Kb);
  gemm_plain(XN, wtv_sa, Vb);
  attn_kernel<<<aGrid, 256, 0, stream>>>(Qb, Kb, Vb, ATT, Tsz, /*causal=*/1);
  gemm_proj(ATT, wtp_sa, bp_sa, x, X1);

  // --- cross-attention: x2 = x1 + proj(attn(ln3(x1), ln2(x_enc))) ---
  ln_bf16_kernel<<<M, 256, 0, stream>>>(X1,    g3, be3, XN);
  ln_bf16_kernel<<<M, 256, 0, stream>>>(x_enc, g2, be2, XE);
  gemm_plain(XN, wtq_ca, Qb);
  gemm_plain(XE, wtk_ca, Kb);
  gemm_plain(XE, wtv_ca, Vb);
  attn_kernel<<<aGrid, 256, 0, stream>>>(Qb, Kb, Vb, ATT, Tsz, /*causal=*/0);
  gemm_proj(ATT, wtp_ca, bp_ca, X1, X2);

  // --- FFN: out = x2 + W2(gelu(W1 ln4(x2) + b1)) + b2 ---
  ln_bf16_kernel<<<M, 256, 0, stream>>>(X2, g4, be4, XN);
  gemm_gelu(XN, wt1, b1, Hb);
  gemm_proj(Hb, wt2, b2, X2, (float*)d_out);
}
